// SupervisedContrastiveEntropyLoss_81235011437123
// MI455X (gfx1250) — matrix operand order repaired and checked by reading, not compiled, not run on hardware
//
#include <hip/hip_runtime.h>
#include <hip/hip_bf16.h>
#include <math.h>

// ---------------------------------------------------------------------------
// SupervisedContrastiveEntropyLoss for MI455X (gfx1250, wave32, WMMA)
//
// out = 0.5 * (attractive + repulsive) + 0.5 * ce
//   attractive = mean_{B,D} (f - center[label])^2
//   repulsive  = sum_{i<j, l_i != l_j} relu(0.5 - dist_ij)^2 / (B(B-1)/2)
//   dist_ij from Gram matrix F F^T  (34 GFLOP -> fp32-via-bf16 split WMMA)
//   ce         = mean cross-entropy of classification vs labels
//
// Gram kernel: one wave per 32x32 macro-tile of the upper triangle.
// 2x2 register tiling: each K-block of 32 feeds 12 v_wmma_f32_16x16x32_bf16
// from 16 global_load_b128 (A/B fragments reused across the 2x2 tiles).
// ---------------------------------------------------------------------------

#define Bsz   8192
#define Dsz   256
#define Csz   128
#define EPSV  1e-6f

typedef __attribute__((ext_vector_type(16))) __bf16   v16bf;
typedef __attribute__((ext_vector_type(8)))  float    v8f;
typedef __attribute__((ext_vector_type(4)))  unsigned int   u32x4;
typedef __attribute__((ext_vector_type(4)))  unsigned short u16x4;

union Frag {
    v16bf bf;
    u32x4 u[2];
};

static __device__ __forceinline__ float wave_sum(float v) {
    #pragma unroll
    for (int o = 16; o > 0; o >>= 1) v += __shfl_xor(v, o, 32);
    return v;
}

// round-to-nearest-even float -> bf16 bits
static __device__ __forceinline__ unsigned short f2bf(float x) {
    unsigned int u = __float_as_uint(x);
    unsigned int r = (u + 0x7FFFu + ((u >> 16) & 1u)) >> 16;
    return (unsigned short)r;
}
static __device__ __forceinline__ float bf2f(unsigned short b) {
    return __uint_as_float(((unsigned int)b) << 16);
}

// ---------------------------------------------------------------------------
// Prep: split features into bf16 hi/lo planes; per-row sq-norm, row-sum,
// attractive partials (one wave per row, 8 rows per block, 1024 blocks).
// ---------------------------------------------------------------------------
__global__ __launch_bounds__(256) void prep_kernel(
    const float* __restrict__ feat, const int* __restrict__ labels,
    unsigned short* __restrict__ hiA, unsigned short* __restrict__ loA,
    float* __restrict__ sqn, float* __restrict__ rs,
    float* __restrict__ att_part)
{
    __shared__ float s_att[8];
    const int wave = threadIdx.x >> 5;
    const int lane = threadIdx.x & 31;
    const int row  = blockIdx.x * 8 + wave;

    const float* fr = feat + (size_t)row * Dsz;
    float sq = 0.f, sm = 0.f;

    #pragma unroll
    for (int c = 0; c < 2; ++c) {
        const int base = c * 128 + lane * 4;
        float4 v = *(const float4*)(fr + base);
        u16x4 h, l;
        float vv[4] = {v.x, v.y, v.z, v.w};
        unsigned short hb[4], lb[4];
        #pragma unroll
        for (int t = 0; t < 4; ++t) {
            hb[t] = f2bf(vv[t]);
            float res = vv[t] - bf2f(hb[t]);
            lb[t] = f2bf(res);
            sq += vv[t] * vv[t];
            sm += vv[t];
        }
        h.x = hb[0]; h.y = hb[1]; h.z = hb[2]; h.w = hb[3];
        l.x = lb[0]; l.y = lb[1]; l.z = lb[2]; l.w = lb[3];
        *(u16x4*)(hiA + (size_t)row * Dsz + base) = h;
        *(u16x4*)(loA + (size_t)row * Dsz + base) = l;
    }

    sq = wave_sum(sq);
    sm = wave_sum(sm);

    if (lane == 0) {
        sqn[row] = sq;
        rs[row]  = sm;
        const int l = labels[row];                 // l in [0,128) < D
        const float s = (l & 1) ? -1.f : 1.f;      // center sign
        const float fv = fr[l];
        s_att[wave] = sq - 2.f * s * fv + 1.f;     // ||f||^2 - 2 s f[l] + 1
    }
    __syncthreads();
    if (threadIdx.x == 0) {
        float a = 0.f;
        #pragma unroll
        for (int w = 0; w < 8; ++w) a += s_att[w];
        att_part[blockIdx.x] = a;
    }
}

// ---------------------------------------------------------------------------
// Cross entropy: one wave per row of classification[8192,128].
// ---------------------------------------------------------------------------
__global__ __launch_bounds__(256) void ce_kernel(
    const float* __restrict__ cls, const int* __restrict__ labels,
    float* __restrict__ ce_part)
{
    __shared__ float s_ce[8];
    const int wave = threadIdx.x >> 5;
    const int lane = threadIdx.x & 31;
    const int row  = blockIdx.x * 8 + wave;

    const float* cr = cls + (size_t)row * Csz;
    float4 v = *(const float4*)(cr + lane * 4);

    float m = fmaxf(fmaxf(v.x, v.y), fmaxf(v.z, v.w));
    #pragma unroll
    for (int o = 16; o > 0; o >>= 1) m = fmaxf(m, __shfl_xor(m, o, 32));

    float e = __expf(v.x - m) + __expf(v.y - m) + __expf(v.z - m) + __expf(v.w - m);
    e = wave_sum(e);
    const float lse = m + __logf(e);

    const int l = labels[row];
    float xl = 0.f;
    if (lane == (l >> 2)) {
        const int q = l & 3;
        xl = (q == 0) ? v.x : (q == 1) ? v.y : (q == 2) ? v.z : v.w;
    }
    xl = wave_sum(xl);   // exactly one lane contributed

    if (lane == 0) s_ce[wave] = lse - xl;
    __syncthreads();
    if (threadIdx.x == 0) {
        float a = 0.f;
        #pragma unroll
        for (int w = 0; w < 8; ++w) a += s_ce[w];
        ce_part[blockIdx.x] = a;
    }
}

// ---------------------------------------------------------------------------
// Gram + hinge: one wave per 32x32 macro-tile (2x2 of 16x16 WMMA tiles)
// over the upper triangle. Per K-block of 32: 12 WMMAs from 16 b128 loads.
// ---------------------------------------------------------------------------
__global__ __launch_bounds__(256) void gram_kernel(
    const unsigned short* __restrict__ hiA, const unsigned short* __restrict__ loA,
    const float* __restrict__ sqn, const float* __restrict__ rs,
    const int* __restrict__ labels, float* __restrict__ rep_part)
{
    __shared__ float s_rep[8];
    const int wave = threadIdx.x >> 5;
    const int lane = threadIdx.x & 31;

    // linear macro-tile-pair index p -> (ti <= tj), p = tj(tj+1)/2 + ti
    const long long p = (long long)blockIdx.x * 8 + wave;
    int tj = (int)((sqrtf(8.0f * (float)p + 1.0f) - 1.0f) * 0.5f);
    while ((long long)tj * (tj + 1) / 2 > p) tj--;
    while ((long long)(tj + 1) * (tj + 2) / 2 <= p) tj++;
    const int ti = (int)(p - (long long)tj * (tj + 1) / 2);

    const int i0 = ti * 32, j0 = tj * 32;
    const int m    = lane & 15;
    const int half = lane >> 4;

    const unsigned short* arow_hi[2];
    const unsigned short* arow_lo[2];
    const unsigned short* brow_hi[2];
    const unsigned short* brow_lo[2];
    #pragma unroll
    for (int t = 0; t < 2; ++t) {
        arow_hi[t] = hiA + (size_t)(i0 + t * 16 + m) * Dsz;
        arow_lo[t] = loA + (size_t)(i0 + t * 16 + m) * Dsz;
        brow_hi[t] = hiA + (size_t)(j0 + t * 16 + m) * Dsz;
        brow_lo[t] = loA + (size_t)(j0 + t * 16 + m) * Dsz;
    }

    v8f acc[2][2];
    #pragma unroll
    for (int ii = 0; ii < 2; ++ii)
        #pragma unroll
        for (int jj = 0; jj < 2; ++jj)
            acc[ii][jj] = (v8f){0.f, 0.f, 0.f, 0.f, 0.f, 0.f, 0.f, 0.f};

    for (int k0 = 0; k0 < Dsz; k0 += 32) {
        Frag ahi[2], alo[2], bhi[2], blo[2];
        #pragma unroll
        for (int t = 0; t < 2; ++t) {
            // A 16x32 bf16 layout: half 0 -> K = {0..7, 16..23}; half 1 -> {8..15, 24..31}
            ahi[t].u[0] = *(const u32x4*)(arow_hi[t] + k0 + half * 8);
            ahi[t].u[1] = *(const u32x4*)(arow_hi[t] + k0 + 16 + half * 8);
            alo[t].u[0] = *(const u32x4*)(arow_lo[t] + k0 + half * 8);
            alo[t].u[1] = *(const u32x4*)(arow_lo[t] + k0 + 16 + half * 8);
            // B 32x16 bf16 layout: column n = lane&15; half 0 -> K = {0..7, 16..23}; half 1 -> {8..15, 24..31}
            bhi[t].u[0] = *(const u32x4*)(brow_hi[t] + k0 + half * 8);
            bhi[t].u[1] = *(const u32x4*)(brow_hi[t] + k0 + 16 + half * 8);
            blo[t].u[0] = *(const u32x4*)(brow_lo[t] + k0 + half * 8);
            blo[t].u[1] = *(const u32x4*)(brow_lo[t] + k0 + 16 + half * 8);
        }
        #pragma unroll
        for (int ii = 0; ii < 2; ++ii) {
            #pragma unroll
            for (int jj = 0; jj < 2; ++jj) {
                acc[ii][jj] = __builtin_amdgcn_wmma_f32_16x16x32_bf16(
                    false, ahi[ii].bf, false, bhi[jj].bf, (short)0, acc[ii][jj], false, false);
                acc[ii][jj] = __builtin_amdgcn_wmma_f32_16x16x32_bf16(
                    false, ahi[ii].bf, false, blo[jj].bf, (short)0, acc[ii][jj], false, false);
                acc[ii][jj] = __builtin_amdgcn_wmma_f32_16x16x32_bf16(
                    false, alo[ii].bf, false, bhi[jj].bf, (short)0, acc[ii][jj], false, false);
            }
        }
    }

    // Epilogue: C/D layout -> VGPR r, lane<16: (M=r, N=lane); lane>=16: (M=r+8, N=lane-16)
    float local = 0.f;
    #pragma unroll
    for (int jj = 0; jj < 2; ++jj) {
        const int j = j0 + jj * 16 + m;
        const float sqn_j = sqn[j];
        const float rs_j  = rs[j];
        const int   lab_j = labels[j];
        #pragma unroll
        for (int ii = 0; ii < 2; ++ii) {
            #pragma unroll
            for (int r = 0; r < 8; ++r) {
                const int i = i0 + ii * 16 + half * 8 + r;
                const float g = acc[ii][jj][r];
                float sq = sqn[i] + sqn_j - 2.f * g
                         + 2.f * EPSV * (rs[i] - rs_j)
                         + (float)Dsz * (EPSV * EPSV);
                const float dist = sqrtf(fmaxf(sq, 1e-12f));
                const float h = fmaxf(0.5f - dist, 0.f);
                const bool ok = (i < j) && (labels[i] != lab_j);
                local += ok ? h * h : 0.f;
            }
        }
    }
    local = wave_sum(local);

    if (lane == 0) s_rep[wave] = local;
    __syncthreads();
    if (threadIdx.x == 0) {
        float a = 0.f;
        #pragma unroll
        for (int w = 0; w < 8; ++w) a += s_rep[w];
        rep_part[blockIdx.x] = a;
    }
}

// ---------------------------------------------------------------------------
// Final deterministic reduction (single block, fixed tree order).
// ---------------------------------------------------------------------------
__global__ __launch_bounds__(256) void finalize_kernel(
    const float* __restrict__ rep_part, int nrep,
    const float* __restrict__ ce_part, const float* __restrict__ att_part, int npart,
    float* __restrict__ out)
{
    __shared__ float sr[256], sc[256], sa[256];
    const int t = threadIdx.x;
    float r = 0.f, c = 0.f, a = 0.f;
    for (int i = t; i < nrep; i += 256) r += rep_part[i];
    for (int i = t; i < npart; i += 256) { c += ce_part[i]; a += att_part[i]; }
    sr[t] = r; sc[t] = c; sa[t] = a;
    __syncthreads();
    for (int o = 128; o > 0; o >>= 1) {
        if (t < o) { sr[t] += sr[t + o]; sc[t] += sc[t + o]; sa[t] += sa[t + o]; }
        __syncthreads();
    }
    if (t == 0) {
        const float attractive = sa[0] / ((float)Bsz * (float)Dsz);
        const float ce         = sc[0] / (float)Bsz;
        const float repulsive  = sr[0] / ((float)Bsz * (float)(Bsz - 1) * 0.5f);
        out[0] = 0.5f * (attractive + repulsive) + 0.5f * ce;
    }
}

// ---------------------------------------------------------------------------
// Launch
// ---------------------------------------------------------------------------
extern "C" void kernel_launch(void* const* d_in, const int* in_sizes, int n_in,
                              void* d_out, int out_size, void* d_ws, size_t ws_size,
                              hipStream_t stream) {
    const float* features       = (const float*)d_in[0];   // [8192,256]
    const float* classification = (const float*)d_in[1];   // [8192,128]
    const int*   labels         = (const int*)d_in[2];     // [8192]
    float*       out            = (float*)d_out;

    // workspace layout (bytes)
    char* ws = (char*)d_ws;
    const size_t HI_OFF  = 0;
    const size_t LO_OFF  = HI_OFF + (size_t)Bsz * Dsz * 2;      // 4 MB
    const size_t SQN_OFF = LO_OFF + (size_t)Bsz * Dsz * 2;      // 8 MB
    const size_t RS_OFF  = SQN_OFF + (size_t)Bsz * 4;
    const size_t ATT_OFF = RS_OFF + (size_t)Bsz * 4;
    const size_t CE_OFF  = ATT_OFF + 1024 * 4;
    const size_t REP_OFF = CE_OFF + 1024 * 4;

    unsigned short* hiA = (unsigned short*)(ws + HI_OFF);
    unsigned short* loA = (unsigned short*)(ws + LO_OFF);
    float* sqn      = (float*)(ws + SQN_OFF);
    float* rs       = (float*)(ws + RS_OFF);
    float* att_part = (float*)(ws + ATT_OFF);
    float* ce_part  = (float*)(ws + CE_OFF);
    float* rep_part = (float*)(ws + REP_OFF);

    const int NMT   = Bsz / 32;                         // 256 macro tiles / dim
    const int NPAIR = NMT * (NMT + 1) / 2;              // 32896 macro-tile pairs
    const int GRAM_BLOCKS = NPAIR / 8;                  // 4112 (8 waves/block)

    prep_kernel<<<Bsz / 8, 256, 0, stream>>>(features, labels, hiA, loA,
                                             sqn, rs, att_part);
    ce_kernel<<<Bsz / 8, 256, 0, stream>>>(classification, labels, ce_part);
    gram_kernel<<<GRAM_BLOCKS, 256, 0, stream>>>(hiA, loA, sqn, rs, labels, rep_part);
    finalize_kernel<<<1, 256, 0, stream>>>(rep_part, GRAM_BLOCKS,
                                           ce_part, att_part, Bsz / 8, out);
}
